// DPQNetwork_60679297958186
// MI455X (gfx1250) — compile-verified, matrix operand order checked by reading, not run on hardware
//
#include <hip/hip_runtime.h>

typedef __attribute__((ext_vector_type(2)))  float    v2f;
typedef __attribute__((ext_vector_type(8)))  float    v8f;
typedef __attribute__((ext_vector_type(16))) _Float16 v16h;

#define BATCH 32768
#define CB    64   // codebooks (C)
#define KC    16   // centroids per codebook (K)
#define DIM   16   // d

// One wave handles a 16-row batch tile x one codebook:
//   D[k, b] = sum_d centroids[c,k,d] * inputs[b,c,d]
// A (SRC0) = centroids tile (M=k, 16x16 f32), B (SRC1) = inputs^T (N=b).
// f32 C/D layout: lane l holds column N=l%16, rows K = 8*(l/16)+r in acc[r].
__global__ __launch_bounds__(256) void dpq_wmma_argmax_kernel(
    const float* __restrict__ inputs,     // (B, C, d)
    const float* __restrict__ centroids,  // (C, K, d)
    int*   __restrict__ codes,            // (B, C) int32
    float* __restrict__ mse)              // (B, C) f32
{
  const int lane = threadIdx.x & 31;
  const int wave = threadIdx.x >> 5;   // 0..7
  const int row  = lane & 15;          // M / N index within tile
  const int half = lane >> 4;          // 0 or 1 (lane half)
  const int btile = blockIdx.x;        // 0..2047
  const int b = btile * 16 + row;

  for (int i = 0; i < 8; ++i) {
    const int c = wave + 8 * i;        // 8 waves cover consecutive c per step
    v8f acc = {};

#if __has_builtin(__builtin_amdgcn_wmma_f32_16x16x4_f32)
    // A 16x4 f32 layout: lane l holds A[M=l%16, K=2*half+{0,1}] (2 VGPRs).
    // B 4x16 f32 layout: lane l holds B[K=2*half+{0,1}, N=l%16].
    const float* cbase = centroids + (c * KC + row) * DIM + 2 * half;
    const float* ibase = inputs + ((size_t)b * CB + c) * DIM + 2 * half;
#pragma unroll
    for (int j = 0; j < 4; ++j) {
      v2f a  = *(const v2f*)(cbase + 4 * j);
      v2f bb = __builtin_nontemporal_load((const v2f*)(ibase + 4 * j));
      acc = __builtin_amdgcn_wmma_f32_16x16x4_f32(
          /*neg_a=*/false, a, /*neg_b=*/false, bb,
          /*c_mod=*/(short)0, acc, /*reuse_a=*/false, /*reuse_b=*/false);
    }
#else
    // Fallback: single zero-padded f16 WMMA (16x16x32, only K<16 used).
    v16h ah, bh;
#pragma unroll
    for (int t = 0; t < 16; ++t) { ah[t] = (_Float16)0.f; bh[t] = (_Float16)0.f; }
    const float* crow = centroids + (c * KC + row) * DIM + 8 * half;
    const float* irow = inputs + ((size_t)b * CB + c) * DIM;
#pragma unroll
    for (int t = 0; t < 8; ++t) ah[t] = (_Float16)crow[t];   // A: K(d)=8*half..+7
    if (half == 0) {
#pragma unroll
      for (int t = 0; t < 16; ++t) bh[t] = (_Float16)irow[t]; // B: rows K=0..15
    }
    acc = __builtin_amdgcn_wmma_f32_16x16x32_f16(
        false, ah, false, bh, (short)0, acc, false, false);
#endif

    // In-lane argmax over this half's 8 K values (strict > keeps lowest idx).
    float best = acc[0];
    int bk = half * 8;
#pragma unroll
    for (int r = 1; r < 8; ++r) {
      float v = acc[r];
      if (v > best) { best = v; bk = half * 8 + r; }
    }

    // Combine the two K-halves: ds_swizzle SWAPX16 (xor=0x10, and=0x1f).
    float obest = __int_as_float(
        __builtin_amdgcn_ds_swizzle(__float_as_int(best), 0x401f));
    int obk = __builtin_amdgcn_ds_swizzle(bk, 0x401f);
    if (obest > best || (obest == best && obk < bk)) { best = obest; bk = obk; }

    if (half == 0) {
      const int oi = b * CB + c;
      codes[oi] = bk + KC * c;  // flattened-codebook offset, per reference
      mse[oi]   = best;
    }
  }
}

extern "C" void kernel_launch(void* const* d_in, const int* in_sizes, int n_in,
                              void* d_out, int out_size, void* d_ws, size_t ws_size,
                              hipStream_t stream) {
  const float* inputs    = (const float*)d_in[0];
  const float* centroids = (const float*)d_in[1];

  // d_out = [codes (B*C int32)] [mse (B*C f32)] [centroids (C*K*d f32)]
  int*   codes    = (int*)d_out;
  float* mse      = (float*)d_out + (size_t)BATCH * CB;
  float* cent_out = (float*)d_out + 2 * (size_t)BATCH * CB;

  dpq_wmma_argmax_kernel<<<BATCH / 16, 256, 0, stream>>>(inputs, centroids,
                                                         codes, mse);
  // tie_in_n_out=True: centroids returned as-is (graph-capture-safe D2D copy).
  hipMemcpyAsync(cent_out, centroids, (size_t)CB * KC * DIM * sizeof(float),
                 hipMemcpyDeviceToDevice, stream);
}
